// Decoder_14963666059264
// MI455X (gfx1250) — matrix operand order repaired and checked by reading, not compiled, not run on hardware
//
#include <hip/hip_runtime.h>
#include <hip/hip_bf16.h>
#include <math.h>

// Problem sizes (match reference)
#define BB 16
#define SS 128
#define EE 256
#define HH 512
#define LL 128          // LSRC
#define VV 32000
#define G4H (4*HH)      // 2048

typedef _Float16 half16 __attribute__((ext_vector_type(16)));
typedef _Float16 half8  __attribute__((ext_vector_type(8)));
typedef float    f32x8  __attribute__((ext_vector_type(8)));

// ---------------------------------------------------------------------------
// WMMA fragment helpers (CDNA5 wave32 layouts, 16x16x32 f16 -> f32)
// A (16x32, row-major src): lane l<16 holds M=l, K={0..7,16..23}; lanes 16..31
// hold M=l-16, K={8..15,24..31}.  Two 16B contiguous loads per lane.
// ---------------------------------------------------------------------------
__device__ inline half16 load_a_frag(const _Float16* __restrict__ p, int ld,
                                     int m0, int k0, int lane) {
    int m  = m0 + (lane & 15);
    int ks = (lane >> 4) * 8;
    half8 lo = *(const half8*)(p + (size_t)m * ld + k0 + ks);
    half8 hi = *(const half8*)(p + (size_t)m * ld + k0 + 16 + ks);
    half16 r;
#pragma unroll
    for (int j = 0; j < 8; ++j) { r[j] = lo[j]; r[j + 8] = hi[j]; }
    return r;
}

// B (32x16, row-major K x N src): lane l holds column N=l%16, K={0..7,16..23}
// for l<16 and K={8..15,24..31} for l>=16 (same K order as A).
__device__ inline half16 load_b_frag(const _Float16* __restrict__ p, int ld,
                                     int k0, int n0, int lane) {
    int n  = n0 + (lane & 15);
    int ks = (lane >> 4) * 8;
    half16 r;
#pragma unroll
    for (int j = 0; j < 8; ++j) {
        r[j]     = p[(size_t)(k0 + ks + j) * ld + n];
        r[j + 8] = p[(size_t)(k0 + 16 + ks + j) * ld + n];
    }
    return r;
}

#define WMMA(a, b, c) \
    __builtin_amdgcn_wmma_f32_16x16x32_f16(false, (a), false, (b), (short)0, (c), false, false)

__device__ inline float sigmoidf_(float x) { return 1.0f / (1.0f + __expf(-x)); }

// ---------------------------------------------------------------------------
// K1a: embedding gather -> f16 X [B*S, E]
// ---------------------------------------------------------------------------
__global__ void embed_kernel(const int* __restrict__ y,
                             const float* __restrict__ tab,
                             _Float16* __restrict__ X) {
    int i = blockIdx.x * blockDim.x + threadIdx.x;   // over B*S*E
    if (i >= BB * SS * EE) return;
    int row = i / EE, e = i - row * EE;
    X[i] = (_Float16)tab[(size_t)y[row] * EE + e];
}

// K1b: W_ih -> f16 transposed + gate-interleaved: WihT[k][n], n = 4*unit+gate
__global__ void conv_wih_kernel(const float* __restrict__ W, _Float16* __restrict__ Wt) {
    int i = blockIdx.x * blockDim.x + threadIdx.x;   // over E*4H
    if (i >= EE * G4H) return;
    int k = i / G4H, n = i - k * G4H;
    int orig = (n & 3) * HH + (n >> 2);              // gate-major original row
    Wt[i] = (_Float16)W[(size_t)orig * EE + k];
}

// K1c: W_hh -> f16 transposed + gate-interleaved: WhhT[k][n]
__global__ void conv_whh_kernel(const float* __restrict__ W, _Float16* __restrict__ Wt) {
    int i = blockIdx.x * blockDim.x + threadIdx.x;   // over H*4H
    if (i >= HH * G4H) return;
    int k = i / G4H, n = i - k * G4H;
    int orig = (n & 3) * HH + (n >> 2);
    Wt[i] = (_Float16)W[(size_t)orig * HH + k];
}

// K1d: fused_enc -> f16 row-major copy + f16 transposed copy (per batch)
__global__ void conv_enc_kernel(const float* __restrict__ enc,
                                _Float16* __restrict__ encF, _Float16* __restrict__ encT) {
    int i = blockIdx.x * blockDim.x + threadIdx.x;   // over B*L*H
    if (i >= BB * LL * HH) return;
    int b = i / (LL * HH), r = i - b * (LL * HH);
    int l = r / HH, h = r - l * HH;
    float v = enc[i];
    encF[i] = (_Float16)v;
    encT[((size_t)b * HH + h) * LL + l] = (_Float16)v;
}

// K1e: W_out [V, 2H] f32 -> WoutT [2H, V] f16, LDS-tiled transpose so both
// sides stay coalesced.  One 32x32 (k x n) tile per block.
__global__ __launch_bounds__(256)
void conv_wout_kernel(const float* __restrict__ W, _Float16* __restrict__ Wt) {
    __shared__ _Float16 tile[32][33];
    int bk = blockIdx.x & 31;            // 1024/32 k-tiles
    int bn = blockIdx.x >> 5;            // 32000/32 n-tiles
    int k0 = bk * 32, n0 = bn * 32;
    int tid = threadIdx.x;
    int tk = tid & 31, tn = tid >> 5;    // read: 32 k consecutive per n-row
#pragma unroll
    for (int r = 0; r < 4; ++r) {
        int n = tn + r * 8;
        tile[tk][n] = (_Float16)W[(size_t)(n0 + n) * (2 * HH) + k0 + tk];
    }
    __syncthreads();
    int tn2 = tid & 31, tk2 = tid >> 5;  // write: 32 n consecutive per k-row
#pragma unroll
    for (int r = 0; r < 4; ++r) {
        int k = tk2 + r * 8;
        Wt[(size_t)(k0 + k) * VV + n0 + tn2] = tile[k][tn2];
    }
}

// ---------------------------------------------------------------------------
// K2: pre-gates G[B*S, 4H] = X @ WihT + (b_ih + b_hh)   (gate-interleaved N)
// one wave per 16x16 tile; grid covers 128 x 128 tiles.
// ---------------------------------------------------------------------------
__global__ __launch_bounds__(256)
void input_gemm_kernel(const _Float16* __restrict__ X, const _Float16* __restrict__ WihT,
                       const float* __restrict__ b_ih, const float* __restrict__ b_hh,
                       float* __restrict__ G) {
    int gw   = (blockIdx.x * blockDim.x + threadIdx.x) >> 5;
    int lane = threadIdx.x & 31;
    int mt = gw >> 7;            // 0..127
    int nt = gw & 127;           // 0..127
    int m0 = mt * 16, n0 = nt * 16;
    int n    = n0 + (lane & 15);
    int orig = (n & 3) * HH + (n >> 2);
    float bias = b_ih[orig] + b_hh[orig];
    f32x8 acc;
#pragma unroll
    for (int v = 0; v < 8; ++v) acc[v] = bias;
#pragma unroll
    for (int k0 = 0; k0 < EE; k0 += 32) {
        half16 a = load_a_frag(X, EE, m0, k0, lane);
        half16 b = load_b_frag(WihT, G4H, k0, n0, lane);
        acc = WMMA(a, b, acc);
    }
    int mb = m0 + ((lane >> 4) * 8);
#pragma unroll
    for (int v = 0; v < 8; ++v)
        G[(size_t)(mb + v) * G4H + n] = acc[v];
}

// ---------------------------------------------------------------------------
// K3: persistent recurrent LSTM. One workgroup (32 waves) on one WGP.
// h (f16) and c (f32) live in LDS. Per step: gates = G[:,s,:] + h @ WhhT,
// then the cell nonlinearity per gate-interleaved tile.
// ---------------------------------------------------------------------------
__global__ __launch_bounds__(1024, 1)
void lstm_kernel(const _Float16* __restrict__ WhhT, const float* __restrict__ G,
                 _Float16* __restrict__ hs) {
    __shared__ __align__(16) _Float16 h_lds[BB * HH];    // 16 KB
    __shared__ float c_lds[BB * HH];                     // 32 KB
    __shared__ float gt[32][256];                        // per-wave 16x16 tile, 32 KB

    int tid = threadIdx.x, wave = tid >> 5, lane = tid & 31;
    for (int i = tid; i < BB * HH; i += 1024) { h_lds[i] = (_Float16)0.0f; c_lds[i] = 0.0f; }
    __syncthreads();

    for (int s = 0; s < SS; ++s) {
        f32x8 acc[4];
#pragma unroll
        for (int t = 0; t < 4; ++t) {
            int n0 = (wave * 4 + t) * 16;
            int n  = n0 + (lane & 15);
#pragma unroll
            for (int v = 0; v < 8; ++v) {
                int m = v + ((lane >> 4) * 8);           // batch index
                acc[t][v] = G[((size_t)m * SS + s) * G4H + n];
            }
            for (int k0 = 0; k0 < HH; k0 += 32) {
                half16 a = load_a_frag(h_lds, HH, 0, k0, lane);
                half16 b = load_b_frag(WhhT, G4H, k0, n0, lane);
                acc[t] = WMMA(a, b, acc[t]);
            }
        }
        // cell update, tile by tile (barriers make cross-lane LDS use safe and
        // also guarantee all h reads completed before h is overwritten)
#pragma unroll
        for (int t = 0; t < 4; ++t) {
            int n0 = (wave * 4 + t) * 16;
            int nl = lane & 15, mb = (lane >> 4) * 8;
#pragma unroll
            for (int v = 0; v < 8; ++v) gt[wave][(mb + v) * 16 + nl] = acc[t][v];
            __syncthreads();
            for (int p = lane; p < 64; p += 32) {
                int m = p >> 2, jl = p & 3;
                float gi = gt[wave][m * 16 + jl * 4 + 0];
                float gf = gt[wave][m * 16 + jl * 4 + 1];
                float gg = gt[wave][m * 16 + jl * 4 + 2];
                float go = gt[wave][m * 16 + jl * 4 + 3];
                int j   = (n0 >> 2) + jl;                // global hidden unit
                int idx = m * HH + j;
                float cn = sigmoidf_(gf) * c_lds[idx] + sigmoidf_(gi) * tanhf(gg);
                float hn = sigmoidf_(go) * tanhf(cn);
                c_lds[idx] = cn;
                h_lds[idx] = (_Float16)hn;
                hs[((size_t)m * SS + s) * HH + j] = (_Float16)hn;
            }
            __syncthreads();
        }
    }
}

// ---------------------------------------------------------------------------
// K4: attention per (batch, 16-row S tile).  scores -> masked softmax ->
// context; fused with the [dec_out | context] concat write (f16).
// ---------------------------------------------------------------------------
__global__ __launch_bounds__(128)
void attention_kernel(const _Float16* __restrict__ hs, const _Float16* __restrict__ encF,
                      const _Float16* __restrict__ encT, const unsigned char* __restrict__ mask,
                      _Float16* __restrict__ outcat) {
    int b = blockIdx.x >> 3, st = blockIdx.x & 7;
    int tid = threadIdx.x, wave = tid >> 5, lane = tid & 31;
    __shared__ float sc[16 * LL];                        // 8 KB
    __shared__ __align__(16) _Float16 pr[16 * LL];       // 4 KB

    const _Float16* Q  = hs   + ((size_t)b * SS + st * 16) * HH;   // [16,H]
    const _Float16* KT = encT + (size_t)b * HH * LL;               // [H,L]
    const _Float16* Ve = encF + (size_t)b * LL * HH;               // [L,H]

    // scores: 8 N-tiles of 16 over LSRC, 2 per wave
#pragma unroll
    for (int u = 0; u < 2; ++u) {
        int n0 = (wave * 2 + u) * 16;
        f32x8 acc = {};
        for (int k0 = 0; k0 < HH; k0 += 32) {
            half16 a  = load_a_frag(Q, HH, 0, k0, lane);
            half16 bb = load_b_frag(KT, LL, k0, n0, lane);
            acc = WMMA(a, bb, acc);
        }
        int n = n0 + (lane & 15), mb = (lane >> 4) * 8;
#pragma unroll
        for (int v = 0; v < 8; ++v) sc[(mb + v) * LL + n] = acc[v];
    }
    __syncthreads();

    // masked, scaled softmax: one row per thread (16 rows)
    if (tid < 16) {
        const float scale = 0.04419417382415922f;        // 1/sqrt(512)
        float mx = -1e30f;
        for (int l = 0; l < LL; ++l) {
            float v = mask[b * LL + l] ? sc[tid * LL + l] * scale : -1e9f;
            sc[tid * LL + l] = v;
            mx = fmaxf(mx, v);
        }
        float sum = 0.0f;
        for (int l = 0; l < LL; ++l) {
            float e = __expf(sc[tid * LL + l] - mx);
            sc[tid * LL + l] = e;
            sum += e;
        }
        float inv = 1.0f / sum;
        for (int l = 0; l < LL; ++l) pr[tid * LL + l] = (_Float16)(sc[tid * LL + l] * inv);
    }
    __syncthreads();

    // context: [16,H] = pr[16,L] @ Ve[L,H]; 32 N-tiles, 8 per wave
#pragma unroll
    for (int u = 0; u < 8; ++u) {
        int n0 = (wave * 8 + u) * 16;
        f32x8 acc = {};
#pragma unroll
        for (int k0 = 0; k0 < LL; k0 += 32) {
            half16 a  = load_a_frag(pr, LL, 0, k0, lane);
            half16 bb = load_b_frag(Ve, HH, k0, n0, lane);
            acc = WMMA(a, bb, acc);
        }
        int n = n0 + (lane & 15), mb = (lane >> 4) * 8;
#pragma unroll
        for (int v = 0; v < 8; ++v) {
            size_t row = (size_t)b * SS + st * 16 + (mb + v);
            outcat[row * (2 * HH) + HH + n] = (_Float16)acc[v];
        }
    }
    // concat first half = dec_out
    for (int i = tid; i < 16 * HH; i += 128) {
        int m = i >> 9, h = i & (HH - 1);
        size_t row = (size_t)b * SS + st * 16 + m;
        outcat[row * (2 * HH) + h] = Q[m * HH + h];
    }
}

// ---------------------------------------------------------------------------
// K5: logits [2048, V] = outcat [2048, 2H] @ WoutT (f16, [2H, V]) + b_out.
// Block = 8 waves handling one 64-row M chunk x 128-col N chunk.  The A chunk
// (64 x 1024 f16 = 128 KB) is staged once in LDS and shared by all 8 waves;
// each wave owns one 16-col N tile and accumulates 4 M tiles in registers,
// so per k-step: 16 strided vmem b16 (B) + 8 ds b128 (A) + 4 WMMA.
// WoutT (64 MB) stays resident in the 192 MB L2 across the whole kernel.
// ---------------------------------------------------------------------------
__global__ __launch_bounds__(256)
void output_gemm_kernel(const _Float16* __restrict__ outcat, const _Float16* __restrict__ WoutT,
                        const float* __restrict__ b_out, float* __restrict__ logits) {
    __shared__ __align__(16) _Float16 Atile[64 * 2 * HH];   // 128 KB
    const int NCH = VV / 128;                // 250 N chunks
    int mc = blockIdx.x / NCH;               // 0..31
    int nc = blockIdx.x % NCH;
    int tid = threadIdx.x, wave = tid >> 5, lane = tid & 31;

    // cooperative stage of the A chunk (coalesced b128)
    const _Float16* Asrc = outcat + (size_t)mc * 64 * (2 * HH);
    for (int i = tid * 8; i < 64 * 2 * HH; i += 256 * 8)
        *(half8*)(Atile + i) = *(const half8*)(Asrc + i);
    __syncthreads();

    int n0 = nc * 128 + wave * 16;
    float bias = b_out[n0 + (lane & 15)];
    f32x8 acc[4];
#pragma unroll
    for (int t = 0; t < 4; ++t)
#pragma unroll
        for (int v = 0; v < 8; ++v) acc[t][v] = bias;

    for (int k0 = 0; k0 < 2 * HH; k0 += 32) {
        half16 bb = load_b_frag(WoutT, VV, k0, n0, lane);
#pragma unroll
        for (int t = 0; t < 4; ++t) {
            half16 a = load_a_frag(Atile, 2 * HH, t * 16, k0, lane);
            acc[t] = WMMA(a, bb, acc[t]);
        }
    }
    int n = n0 + (lane & 15), mb = (lane >> 4) * 8;
#pragma unroll
    for (int t = 0; t < 4; ++t)
#pragma unroll
        for (int v = 0; v < 8; ++v)
            logits[(size_t)(mc * 64 + t * 16 + mb + v) * VV + n] = acc[t][v];
}

// ---------------------------------------------------------------------------
extern "C" void kernel_launch(void* const* d_in, const int* in_sizes, int n_in,
                              void* d_out, int out_size, void* d_ws, size_t ws_size,
                              hipStream_t stream) {
    (void)in_sizes; (void)n_in; (void)out_size; (void)ws_size;
    const int*   y      = (const int*)d_in[0];
    const float* tab    = (const float*)d_in[1];
    const float* W_ih   = (const float*)d_in[2];
    const float* W_hh   = (const float*)d_in[3];
    const float* b_ih   = (const float*)d_in[4];
    const float* b_hh   = (const float*)d_in[5];
    const float* enc    = (const float*)d_in[6];
    const unsigned char* mask = (const unsigned char*)d_in[7];
    const float* W_out  = (const float*)d_in[8];
    const float* b_out  = (const float*)d_in[9];
    float*       logits = (float*)d_out;

    // workspace carve-up (f16 staging + f32 pre-gates), ~95.5 MB total
    char* ws = (char*)d_ws;
    size_t off = 0;
    _Float16* X     = (_Float16*)(ws + off); off += (size_t)BB*SS*EE*2;        // 1 MB
    _Float16* WihT  = (_Float16*)(ws + off); off += (size_t)EE*G4H*2;          // 1 MB
    _Float16* WhhT  = (_Float16*)(ws + off); off += (size_t)HH*G4H*2;          // 2 MB
    _Float16* encF  = (_Float16*)(ws + off); off += (size_t)BB*LL*HH*2;        // 2 MB
    _Float16* encT  = (_Float16*)(ws + off); off += (size_t)BB*HH*LL*2;        // 2 MB
    _Float16* hs    = (_Float16*)(ws + off); off += (size_t)BB*SS*HH*2;        // 2 MB
    _Float16* ocat  = (_Float16*)(ws + off); off += (size_t)BB*SS*2*HH*2;      // 4 MB
    float*    G     = (float*)(ws + off);    off += (size_t)BB*SS*G4H*4;       // 16 MB
    _Float16* WoutT = (_Float16*)(ws + off); off += (size_t)(2*HH)*VV*2;       // 64 MB

    embed_kernel    <<<(BB*SS*EE + 255) / 256, 256, 0, stream>>>(y, tab, X);
    conv_wih_kernel <<<(EE*G4H  + 255) / 256, 256, 0, stream>>>(W_ih, WihT);
    conv_whh_kernel <<<(HH*G4H  + 255) / 256, 256, 0, stream>>>(W_hh, WhhT);
    conv_enc_kernel <<<(BB*LL*HH+ 255) / 256, 256, 0, stream>>>(enc, encF, encT);
    conv_wout_kernel<<<(2*HH/32) * (VV/32), 256, 0, stream>>>(W_out, WoutT);

    // pre-gates: 128 x 128 tiles, 8 waves/block
    input_gemm_kernel<<<(128 * 128) / 8, 256, 0, stream>>>(X, WihT, b_ih, b_hh, G);

    // sequential recurrence: one persistent workgroup
    lstm_kernel<<<1, 1024, 0, stream>>>(WhhT, G, hs);

    // attention + concat: one block per (b, s-tile)
    attention_kernel<<<BB * (SS / 16), 128, 0, stream>>>(hs, encF, encT, mask, ocat);

    // output projection: 32 M-chunks x 250 N-chunks blocks
    output_gemm_kernel<<<32 * (VV / 128), 256, 0, stream>>>(ocat, WoutT, b_out, logits);
}
